// NVFP4Linear_47425028882638
// MI455X (gfx1250) — matrix operand order repaired and checked by reading, not compiled, not run on hardware
//
#include <hip/hip_runtime.h>
#include <hip/hip_bf16.h>

typedef __attribute__((ext_vector_type(16))) _Float16 v16h;
typedef __attribute__((ext_vector_type(8)))  _Float16 v8h;
typedef __attribute__((ext_vector_type(8)))  float    v8f;

#define F4_MAX 6.0f
#define F8_MAX 448.0f
#define TS_DIV (F4_MAX * F8_MAX)   // 2688

// ---- E4M3 rounding (once per 16-block; mirrors reference formula, RNE via rintf) ----
__device__ __forceinline__ float round_e4m3(float x) {
    float a = fminf(fmaxf(x, 0.0f), F8_MAX);
    float safe = fmaxf(a, 1e-30f);
    float e = fminf(fmaxf(floorf(__log2f(safe)), -6.0f), 8.0f);
    float step = exp2f(e - 3.0f);
    float q = rintf(a / step) * step;
    return (a == 0.0f) ? 0.0f : fminf(fmaxf(q, 0.0f), F8_MAX);
}

__device__ __forceinline__ float tensor_scale(float amax) {
    float t = amax / TS_DIV;
    return (t == 0.0f) ? 1.0f : t;
}

// ---- NVFP4 quant of one 16-block -> f16 (q * bscale); E2M1 via RNE comparison ladder ----
// Tie points of round_e2m1 under RNE: 0.25->0, 0.75->1, 1.25->1, 1.75->2, 2.5->2, 3.5->4, 5->4.
__device__ __forceinline__ void nvfp4_quant16(const float v[16], float ts, v8h& h0, v8h& h1) {
    float bm = 0.0f;
#pragma unroll
    for (int j = 0; j < 16; ++j) bm = fmaxf(bm, fabsf(v[j]));
    float bs = round_e4m3(bm / (F4_MAX * ts));
    float dn = bs * ts;
    float ds = (dn == 0.0f) ? 1.0f : dn;
    const float T0 = 0.25f * ds, T1 = 0.75f * ds, T2 = 1.25f * ds, T3 = 1.75f * ds;
    const float T4 = 2.50f * ds, T5 = 3.50f * ds, T6 = 5.00f * ds;
#pragma unroll
    for (int j = 0; j < 8; ++j) {
        float a = fabsf(v[j]);
        float q = (a <= T0) ? 0.0f : (a < T1) ? 0.5f : (a <= T2) ? 1.0f : (a < T3) ? 1.5f
                : (a <= T4) ? 2.0f : (a < T5) ? 3.0f : (a <= T6) ? 4.0f : 6.0f;
        h0[j] = (_Float16)copysignf(q * bs, v[j]);
    }
#pragma unroll
    for (int j = 0; j < 8; ++j) {
        float a = fabsf(v[8 + j]);
        float q = (a <= T0) ? 0.0f : (a < T1) ? 0.5f : (a <= T2) ? 1.0f : (a < T3) ? 1.5f
                : (a <= T4) ? 2.0f : (a < T5) ? 3.0f : (a <= T6) ? 4.0f : 6.0f;
        h1[j] = (_Float16)copysignf(q * bs, v[8 + j]);
    }
}

// ---- kernel 0: zero amax slots ----
__global__ void init_ws_kernel(unsigned* slots) {
    if (threadIdx.x < 2) slots[threadIdx.x] = 0u;
}

// ---- kernel 1: per-tensor abs-max ----
__global__ void amax_kernel(const float* __restrict__ p, long n4, unsigned* __restrict__ slot) {
    __shared__ float red[256];
    const float4* p4 = (const float4*)p;
    float m = 0.0f;
    for (long i = (long)blockIdx.x * blockDim.x + threadIdx.x; i < n4;
         i += (long)gridDim.x * blockDim.x) {
        float4 v = p4[i];
        m = fmaxf(m, fmaxf(fmaxf(fabsf(v.x), fabsf(v.y)), fmaxf(fabsf(v.z), fabsf(v.w))));
    }
    red[threadIdx.x] = m;
    __syncthreads();
    for (int s = 128; s > 0; s >>= 1) {
        if ((int)threadIdx.x < s) red[threadIdx.x] = fmaxf(red[threadIdx.x], red[threadIdx.x + s]);
        __syncthreads();
    }
    if (threadIdx.x == 0) atomicMax(slot, __float_as_uint(red[0]));
}

// ---- kernel 2: quantize a tensor (one thread per 16-block) -> f16 (q*bscale) table ----
__global__ void quant_tensor_kernel(const float* __restrict__ src, _Float16* __restrict__ dst,
                                    const unsigned* __restrict__ amax, int slot, long nblk) {
    long t = (long)blockIdx.x * 256 + threadIdx.x;
    if (t >= nblk) return;
    float ts = tensor_scale(__uint_as_float(amax[slot]));
    const float4* p = (const float4*)(src + t * 16);
    float v[16];
#pragma unroll
    for (int j = 0; j < 4; ++j) {
        float4 q = p[j];
        v[4*j] = q.x; v[4*j+1] = q.y; v[4*j+2] = q.z; v[4*j+3] = q.w;
    }
    v8h h0, h1;
    nvfp4_quant16(v, ts, h0, h1);
    *(v8h*)(dst + t * 16)     = h0;
    *(v8h*)(dst + t * 16 + 8) = h1;
}

// ---- kernel 3 (primary): f16 WMMA GEMM over prequantized xq/wq, double-buffered LDS ----
// block = 128x128 tile, 256 threads (8 waves), BK=64; wave -> 32x64 patch (2x4 wmma tiles)
__global__ __launch_bounds__(256)
void gemm_pre_kernel(const _Float16* __restrict__ xq, const _Float16* __restrict__ wq,
                     const float* __restrict__ bias, float* __restrict__ out,
                     const unsigned* __restrict__ amax, int K, int N) {
    constexpr int BM = 128, BN = 128, BK = 64, LDT = 72;
    __shared__ __align__(16) _Float16 sx[2][BM][LDT];
    __shared__ __align__(16) _Float16 sw[2][BN][LDT];

    const int m0 = blockIdx.y * BM;
    const int n0 = blockIdx.x * BN;
    const int tid = threadIdx.x;

    const float sglobal = tensor_scale(__uint_as_float(amax[0])) *
                          tensor_scale(__uint_as_float(amax[1]));

    const int wv = tid >> 5, lane = tid & 31;
    const int wr = wv & 3, wc = wv >> 2;
    const int lm = lane & 15, kh = lane >> 4;

    v8f acc[2][4];
#pragma unroll
    for (int mi = 0; mi < 2; ++mi)
#pragma unroll
        for (int nj = 0; nj < 4; ++nj) acc[mi][nj] = (v8f){};

    v8h rx[4], rw[4];
    auto issue = [&](int kt) {
#pragma unroll
        for (int it = 0; it < 4; ++it) {
            int c = tid + it * 256;            // 1024 chunks: 128 rows x 8 x (8 halves)
            int row = c >> 3, cc = c & 7;
            rx[it] = *(const v8h*)(xq + (size_t)(m0 + row) * K + kt + cc * 8);
            rw[it] = *(const v8h*)(wq + (size_t)(n0 + row) * K + kt + cc * 8);
        }
    };
    auto commit = [&](int buf) {
#pragma unroll
        for (int it = 0; it < 4; ++it) {
            int c = tid + it * 256;
            int row = c >> 3, cc = c & 7;
            *(v8h*)&sx[buf][row][cc * 8] = rx[it];
            *(v8h*)&sw[buf][row][cc * 8] = rw[it];
        }
    };

    issue(0);
    commit(0);
    __syncthreads();

    const int nk = K / BK;
    for (int t = 0; t < nk; ++t) {
        const int cur = t & 1;
        if (t + 1 < nk) issue((t + 1) * BK);   // global loads overlap the WMMAs below

#pragma unroll
        for (int kk = 0; kk < BK; kk += 32) {
            v16h a[2], b[4];
#pragma unroll
            for (int mi = 0; mi < 2; ++mi) {
                int row = wr * 32 + mi * 16 + lm;   // A: lane = M row; K halves at lane>=16
                v8h lo = *(const v8h*)&sx[cur][row][kk + kh * 16];
                v8h hi = *(const v8h*)&sx[cur][row][kk + kh * 16 + 8];
                a[mi] = __builtin_shufflevector(lo, hi,
                        0,1,2,3,4,5,6,7,8,9,10,11,12,13,14,15);
            }
#pragma unroll
            for (int nj = 0; nj < 4; ++nj) {
                int row = wc * 64 + nj * 16 + lm;   // B: lane = N col; K 0-15/16-31 halves
                v8h lo = *(const v8h*)&sw[cur][row][kk + kh * 16];
                v8h hi = *(const v8h*)&sw[cur][row][kk + kh * 16 + 8];
                b[nj] = __builtin_shufflevector(lo, hi,
                        0,1,2,3,4,5,6,7,8,9,10,11,12,13,14,15);
            }
#pragma unroll
            for (int mi = 0; mi < 2; ++mi)
#pragma unroll
                for (int nj = 0; nj < 4; ++nj)
                    acc[mi][nj] = __builtin_amdgcn_wmma_f32_16x16x32_f16(
                        false, a[mi], false, b[nj], (short)0, acc[mi][nj], false, false);
        }

        if (t + 1 < nk) commit(cur ^ 1);
        __syncthreads();
    }

#pragma unroll
    for (int mi = 0; mi < 2; ++mi) {
#pragma unroll
        for (int nj = 0; nj < 4; ++nj) {
            int col = n0 + wc * 64 + nj * 16 + lm;
            float bb = bias[col];
            int mrow = m0 + wr * 32 + mi * 16 + kh * 8;   // D: VGPR i -> M = i + 8*kh
#pragma unroll
            for (int i = 0; i < 8; ++i)
                __builtin_nontemporal_store(acc[mi][nj][i] * sglobal + bb,
                                            &out[(size_t)(mrow + i) * N + col]);
        }
    }
}

// ---- kernel 3' (fallback, small ws): fused on-the-fly x quant + WMMA GEMM ----
__global__ __launch_bounds__(256)
void gemm_fused_kernel(const float* __restrict__ x, const _Float16* __restrict__ wq,
                       const float* __restrict__ bias, float* __restrict__ out,
                       const unsigned* __restrict__ amax, int K, int N) {
    constexpr int BM = 128, BN = 128, BK = 64, LDT = 72;
    __shared__ __align__(16) _Float16 sx[BM][LDT];
    __shared__ __align__(16) _Float16 sw[BN][LDT];

    const int m0 = blockIdx.y * BM;
    const int n0 = blockIdx.x * BN;
    const int tid = threadIdx.x;

    const float tsx = tensor_scale(__uint_as_float(amax[0]));
    const float tsw = tensor_scale(__uint_as_float(amax[1]));
    const float sglobal = tsx * tsw;

    const int wv = tid >> 5, lane = tid & 31;
    const int wr = wv & 3, wc = wv >> 2;
    const int lm = lane & 15, kh = lane >> 4;

    v8f acc[2][4];
#pragma unroll
    for (int mi = 0; mi < 2; ++mi)
#pragma unroll
        for (int nj = 0; nj < 4; ++nj) acc[mi][nj] = (v8f){};

    for (int kt = 0; kt < K; kt += BK) {
#pragma unroll
        for (int it = 0; it < 2; ++it) {
            int blk = tid + it * 256;
            int row = blk >> 2, bc = blk & 3;
            const float4* px = (const float4*)(x + (size_t)(m0 + row) * K + kt + bc * 16);
            float v[16];
#pragma unroll
            for (int j = 0; j < 4; ++j) {
                float4 q = px[j];
                v[4*j] = q.x; v[4*j+1] = q.y; v[4*j+2] = q.z; v[4*j+3] = q.w;
            }
            v8h h0, h1;
            nvfp4_quant16(v, tsx, h0, h1);
            *(v8h*)&sx[row][bc * 16]     = h0;
            *(v8h*)&sx[row][bc * 16 + 8] = h1;
        }
#pragma unroll
        for (int it = 0; it < 4; ++it) {
            int c = tid + it * 256;
            int row = c >> 3, cc = c & 7;
            *(v8h*)&sw[row][cc * 8] = *(const v8h*)(wq + (size_t)(n0 + row) * K + kt + cc * 8);
        }
        __syncthreads();

#pragma unroll
        for (int kk = 0; kk < BK; kk += 32) {
            v16h a[2], b[4];
#pragma unroll
            for (int mi = 0; mi < 2; ++mi) {
                int row = wr * 32 + mi * 16 + lm;
                v8h lo = *(const v8h*)&sx[row][kk + kh * 16];
                v8h hi = *(const v8h*)&sx[row][kk + kh * 16 + 8];
                a[mi] = __builtin_shufflevector(lo, hi,
                        0,1,2,3,4,5,6,7,8,9,10,11,12,13,14,15);
            }
#pragma unroll
            for (int nj = 0; nj < 4; ++nj) {
                int row = wc * 64 + nj * 16 + lm;
                v8h lo = *(const v8h*)&sw[row][kk + kh * 16];
                v8h hi = *(const v8h*)&sw[row][kk + kh * 16 + 8];
                b[nj] = __builtin_shufflevector(lo, hi,
                        0,1,2,3,4,5,6,7,8,9,10,11,12,13,14,15);
            }
#pragma unroll
            for (int mi = 0; mi < 2; ++mi)
#pragma unroll
                for (int nj = 0; nj < 4; ++nj)
                    acc[mi][nj] = __builtin_amdgcn_wmma_f32_16x16x32_f16(
                        false, a[mi], false, b[nj], (short)0, acc[mi][nj], false, false);
        }
        __syncthreads();
    }

#pragma unroll
    for (int mi = 0; mi < 2; ++mi) {
#pragma unroll
        for (int nj = 0; nj < 4; ++nj) {
            int col = n0 + wc * 64 + nj * 16 + lm;
            float bb = bias[col];
            int mrow = m0 + wr * 32 + mi * 16 + kh * 8;
#pragma unroll
            for (int i = 0; i < 8; ++i)
                __builtin_nontemporal_store(acc[mi][nj][i] * sglobal + bb,
                                            &out[(size_t)(mrow + i) * N + col]);
        }
    }
}

extern "C" void kernel_launch(void* const* d_in, const int* in_sizes, int n_in,
                              void* d_out, int out_size, void* d_ws, size_t ws_size,
                              hipStream_t stream) {
    const float* x    = (const float*)d_in[0];
    const float* w    = (const float*)d_in[1];
    const float* bias = (const float*)d_in[2];
    float* out = (float*)d_out;

    const int  N = in_sizes[2];              // 1024
    const int  K = in_sizes[1] / N;          // 1024
    const long T = (long)in_sizes[0] / K;    // 65536

    unsigned* amax = (unsigned*)d_ws;
    _Float16* wq   = (_Float16*)((char*)d_ws + 16);                       // 2 MB
    _Float16* xq   = (_Float16*)((char*)d_ws + 16 + (size_t)in_sizes[1] * 2);

    const size_t need_full = 16 + (size_t)in_sizes[1] * 2 + (size_t)in_sizes[0] * 2;

    init_ws_kernel<<<1, 32, 0, stream>>>(amax);
    amax_kernel<<<8192, 256, 0, stream>>>(x, (long)in_sizes[0] / 4, amax + 0);
    amax_kernel<<<256, 256, 0, stream>>>(w, (long)in_sizes[1] / 4, amax + 1);

    long nblk_w = (long)in_sizes[1] / 16;
    quant_tensor_kernel<<<(unsigned)((nblk_w + 255) / 256), 256, 0, stream>>>(w, wq, amax, 1, nblk_w);

    dim3 grid(N / 128, (unsigned)(T / 128));
    if (ws_size >= need_full) {
        long nblk_x = (long)in_sizes[0] / 16;
        quant_tensor_kernel<<<(unsigned)((nblk_x + 255) / 256), 256, 0, stream>>>(x, xq, amax, 0, nblk_x);
        gemm_pre_kernel<<<grid, 256, 0, stream>>>(xq, wq, bias, out, amax, K, N);
    } else {
        gemm_fused_kernel<<<grid, 256, 0, stream>>>(x, wq, bias, out, amax, K, N);
    }
}
